// AttentionBlock_601295421822
// MI455X (gfx1250) — compile-verified, matrix operand order checked by reading, not run on hardware
//
#include <hip/hip_runtime.h>
#include <math.h>

typedef unsigned int u32;
typedef __attribute__((ext_vector_type(16))) _Float16 v16h;
typedef __attribute__((ext_vector_type(8)))  float    v8f;
typedef __attribute__((ext_vector_type(8)))  u32      v8u;

#define NN 25000
#define MM 150000
// D = 512, H = 8, DK = DV = 64, QKV row = 1536

// ---------------- elementwise helpers ----------------

__global__ void k_f32_to_f16(const float* __restrict__ src, _Float16* __restrict__ dst, long n) {
    long i = (long)blockIdx.x * blockDim.x + threadIdx.x;
    if (i < n) dst[i] = (_Float16)src[i];
}

// W: [K x Ncols] fp32 row-major -> Wt rows [rowOffset+n][k] f16 (B-transposed, K-major)
__global__ void k_w_transpose_f16(const float* __restrict__ W, _Float16* __restrict__ Wt,
                                  int K, int Ncols, int rowOffset) {
    int i = blockIdx.x * blockDim.x + threadIdx.x;
    if (i >= K * Ncols) return;
    int k = i / Ncols, n = i % Ncols;
    Wt[(long)(rowOffset + n) * K + k] = (_Float16)W[i];
}

__global__ void k_bias_concat(const float* __restrict__ bq, const float* __restrict__ bk,
                              const float* __restrict__ bv, float* __restrict__ bcat) {
    int i = blockIdx.x * blockDim.x + threadIdx.x;
    if (i >= 1536) return;
    bcat[i] = (i < 512) ? bq[i] : (i < 1024) ? bk[i - 512] : bv[i - 1024];
}

// ---------------- WMMA GEMM ----------------
// C[M x Nc] = A[M x K](f16) * Bt[Nc x K](f16)^T + bias.
// blockDim = 128 (4 waves); each wave computes a 16x64 output strip: one A fragment
// reused across 4 WMMAs. Copy-free ping-pong double buffering; the 4 B rows use
// compile-time byte offsets (t*16*KD) folded into the load immediates.

__device__ __forceinline__ void load_fragA(const _Float16* __restrict__ row, int kb, v8u& u) {
#pragma unroll
    for (int v = 0; v < 8; ++v) {
        const int kk = kb + ((v < 4) ? (2 * v) : (16 + 2 * (v - 4)));
        u[v] = *(const u32*)(row + kk);     // two adjacent-K halves, 4B aligned
    }
}

template <int KD>
__device__ __forceinline__ void load_stepB(const _Float16* __restrict__ Brow0, int kb,
                                           v8u (&b)[4]) {
#pragma unroll
    for (int t = 0; t < 4; ++t)
        load_fragA(Brow0 + (long)t * 16 * KD, kb, b[t]);   // const offsets, fit imm24
}

template <int KD>
__global__ void k_gemm_wmma_f16(const _Float16* __restrict__ A, const _Float16* __restrict__ Bt,
                                const float* __restrict__ bias, float* __restrict__ C,
                                int Mrows, int Ncols) {
    const int lane  = threadIdx.x & 31;
    const int wave  = threadIdx.x >> 5;
    const int tileM = blockIdx.x;
    const int nBase = (blockIdx.y * 4 + wave) * 64;   // 64-wide strip per wave
    const int r16   = lane & 15;
    const int hi    = lane >> 4;             // ISA 16-bit A/B layout: lane group -> K halves

    int m = tileM * 16 + r16;
    if (m >= Mrows) m = Mrows - 1;           // clamp; stores guarded below
    const _Float16* Arow  = A  + (long)m * KD;
    const _Float16* Brow0 = Bt + (long)(nBase + r16) * KD;   // Ncols multiple of 256

    v8u a0, a1, b0[4], b1[4];
    v8f acc[4] = {};

    load_fragA(Arow, hi * 8, a0);
    load_stepB<KD>(Brow0, hi * 8, b0);

#pragma unroll 2
    for (int k0 = 0; k0 < KD; k0 += 64) {
        // phase 0: preload k0+32 into buf1, consume buf0 (k0)
        {
            __builtin_prefetch(Arow + k0 + 192, 0, 3);
            const int kb = k0 + 32 + hi * 8;
            load_fragA(Arow, kb, a1);
            load_stepB<KD>(Brow0, kb, b1);
        }
#pragma unroll
        for (int t = 0; t < 4; ++t)
            acc[t] = __builtin_amdgcn_wmma_f32_16x16x32_f16(
                         false, __builtin_bit_cast(v16h, a0),
                         false, __builtin_bit_cast(v16h, b0[t]),
                         (short)0, acc[t], false, false);
        // phase 1: preload k0+64 into buf0, consume buf1 (k0+32)
        if (k0 + 64 < KD) {
            const int kb = k0 + 64 + hi * 8;
            load_fragA(Arow, kb, a0);
            load_stepB<KD>(Brow0, kb, b0);
        }
#pragma unroll
        for (int t = 0; t < 4; ++t)
            acc[t] = __builtin_amdgcn_wmma_f32_16x16x32_f16(
                         false, __builtin_bit_cast(v16h, a1),
                         false, __builtin_bit_cast(v16h, b1[t]),
                         (short)0, acc[t], false, false);
    }

    const int mb = tileM * 16 + hi * 8;      // C layout: M = v + 8*hi, N = lane%16
#pragma unroll
    for (int t = 0; t < 4; ++t) {
        const int col = nBase + t * 16 + r16;
        const float bv = bias[col];
#pragma unroll
        for (int v = 0; v < 8; ++v) {
            const int mr = mb + v;
            if (mr < Mrows) C[(long)mr * Ncols + col] = acc[t][v] + bv;
        }
    }
}

// ---------------- edge scores + global max ----------------

__device__ __forceinline__ u32 f32_ord(float f) {
    u32 u = __float_as_uint(f);
    return (u & 0x80000000u) ? ~u : (u | 0x80000000u);
}
__device__ __forceinline__ float f32_unord(u32 t) {
    return __uint_as_float((t & 0x80000000u) ? (t & 0x7FFFFFFFu) : ~t);
}

// one wave per edge; 4 lanes per head, 16 dims per lane
__global__ void k_edge_scores(const float* __restrict__ QKV, const int* __restrict__ ei,
                              float* __restrict__ score, u32* __restrict__ gmax, int M) {
    const int widx = (int)((blockIdx.x * (long)blockDim.x + threadIdx.x) >> 5);
    const int lane = threadIdx.x & 31;
    float my = -INFINITY;
    if (widx < M) {
        const int s = ei[widx], r = ei[M + widx];
        const int h = lane >> 2, seg = lane & 3;
        const float* q = QKV + (long)r * 1536 + h * 64 + seg * 16;
        const float* k = QKV + (long)s * 1536 + 512 + h * 64 + seg * 16;
        float p = 0.f;
#pragma unroll
        for (int i = 0; i < 16; ++i) p += q[i] * k[i];
        p += __shfl_xor(p, 1);
        p += __shfl_xor(p, 2);
        p *= 0.125f;                         // 1/sqrt(DK)
        if (seg == 0) { score[(long)widx * 8 + h] = p; my = p; }
    }
    __shared__ float smax[8];
    float wm = my;
#pragma unroll
    for (int off = 16; off > 0; off >>= 1) wm = fmaxf(wm, __shfl_xor(wm, off));
    if (lane == 0) smax[threadIdx.x >> 5] = wm;
    __syncthreads();
    if (threadIdx.x < 32) {
        float b = (lane < (int)(blockDim.x >> 5)) ? smax[lane] : -INFINITY;
#pragma unroll
        for (int off = 16; off > 0; off >>= 1) b = fmaxf(b, __shfl_xor(b, off));
        if (lane == 0) atomicMax(gmax, f32_ord(b));
    }
}

// exp shift + segment-sum denominators (score buffer becomes exp(att) in place)
__global__ void k_exp_ssum(float* __restrict__ score, const int* __restrict__ ei,
                           const u32* __restrict__ gmax, float* __restrict__ ssum, int M) {
    const int i = blockIdx.x * blockDim.x + threadIdx.x;
    if (i >= M * 8) return;
    const float g = f32_unord(*gmax);
    const float e = expf(score[i] + 3.0f - g);
    score[i] = e;
    const int m = i >> 3, h = i & 7;
    atomicAdd(&ssum[(long)ei[M + m] * 8 + h], e);
}

// one wave per edge: normalized weighted message scatter-add
__global__ void k_scatter(const float* __restrict__ QKV, const float* __restrict__ expatt,
                          const float* __restrict__ ssum, const int* __restrict__ ei,
                          float* __restrict__ agg, int M) {
    const int widx = (int)((blockIdx.x * (long)blockDim.x + threadIdx.x) >> 5);
    if (widx >= M) return;
    const int lane = threadIdx.x & 31;
    const int s = ei[widx], r = ei[M + widx];
    float w[8];
#pragma unroll
    for (int h = 0; h < 8; ++h) {
        const float den = ssum[(long)r * 8 + h];
        w[h] = (den == 0.0f) ? 0.0f : expatt[(long)widx * 8 + h] / den;
    }
    const float* V = QKV + (long)s * 1536 + 1024;
    float* out     = agg + (long)r * 512;
#pragma unroll
    for (int i = 0; i < 16; ++i) {
        const int idx = lane + i * 32;       // head = idx>>6 == i>>1 (compile-time)
        atomicAdd(&out[idx], w[i >> 1] * V[idx]);
    }
}

// ---------------- launch ----------------

extern "C" void kernel_launch(void* const* d_in, const int* in_sizes, int n_in,
                              void* d_out, int out_size, void* d_ws, size_t ws_size,
                              hipStream_t stream) {
    const float* x    = (const float*)d_in[0];
    const int*   ei   = (const int*)  d_in[1];
    const float* Wk   = (const float*)d_in[2];
    const float* bk   = (const float*)d_in[3];
    const float* Wq   = (const float*)d_in[4];
    const float* bq   = (const float*)d_in[5];
    const float* Wv   = (const float*)d_in[6];
    const float* bv   = (const float*)d_in[7];
    const float* Wff  = (const float*)d_in[8];
    const float* bff  = (const float*)d_in[9];
    float* out = (float*)d_out;

    // workspace carve-out (256B aligned)
    size_t off = 0;
    auto carve = [&](size_t bytes) { size_t o = off; off += (bytes + 255) & ~(size_t)255; return o; };
    char* ws = (char*)d_ws;
    _Float16* xh    = (_Float16*)(ws + carve((size_t)NN * 512 * 2));
    _Float16* WtQKV = (_Float16*)(ws + carve((size_t)1536 * 512 * 2));
    _Float16* WtFF  = (_Float16*)(ws + carve((size_t)512 * 512 * 2));
    float*    bcat  = (float*)   (ws + carve(1536 * 4));
    float*    QKV   = (float*)   (ws + carve((size_t)NN * 1536 * 4));
    float*    score = (float*)   (ws + carve((size_t)MM * 8 * 4));
    float*    ssum  = (float*)   (ws + carve((size_t)NN * 8 * 4));
    float*    agg   = (float*)   (ws + carve((size_t)NN * 512 * 4));
    _Float16* aggh  = (_Float16*)(ws + carve((size_t)NN * 512 * 2));
    u32*      gmax  = (u32*)     (ws + carve(256));

    // zero accumulators (async, capture-safe)
    hipMemsetAsync(ssum, 0, (size_t)NN * 8 * 4, stream);
    hipMemsetAsync(agg,  0, (size_t)NN * 512 * 4, stream);
    hipMemsetAsync(gmax, 0, 4, stream);  // 0 <= ord(x) for every float x: lower bound

    // precision conversion + weight transposes
    {
        long n = (long)NN * 512;
        k_f32_to_f16<<<(int)((n + 255) / 256), 256, 0, stream>>>(x, xh, n);
    }
    {
        int n = 512 * 512, blk = (n + 255) / 256;
        k_w_transpose_f16<<<blk, 256, 0, stream>>>(Wq,  WtQKV, 512, 512, 0);
        k_w_transpose_f16<<<blk, 256, 0, stream>>>(Wk,  WtQKV, 512, 512, 512);
        k_w_transpose_f16<<<blk, 256, 0, stream>>>(Wv,  WtQKV, 512, 512, 1024);
        k_w_transpose_f16<<<blk, 256, 0, stream>>>(Wff, WtFF,  512, 512, 0);
    }
    k_bias_concat<<<6, 256, 0, stream>>>(bq, bk, bv, bcat);

    // GEMM 1: QKV[N x 1536] = x @ [Wq|Wk|Wv] + [bq|bk|bv]
    {
        dim3 grid((NN + 15) / 16, 1536 / 256);
        k_gemm_wmma_f16<512><<<grid, 128, 0, stream>>>(xh, WtQKV, bcat, QKV, NN, 1536);
    }

    // edge attention
    k_edge_scores<<<(MM * 32) / 256, 256, 0, stream>>>(QKV, ei, score, gmax, MM);
    k_exp_ssum<<<(MM * 8 + 255) / 256, 256, 0, stream>>>(score, ei, gmax, ssum, MM);
    k_scatter<<<(MM * 32) / 256, 256, 0, stream>>>(QKV, score, ssum, ei, agg, MM);

    // GEMM 2: out[N x 512] = agg @ Wff + bff
    {
        long n = (long)NN * 512;
        k_f32_to_f16<<<(int)((n + 255) / 256), 256, 0, stream>>>(agg, aggh, n);
        dim3 grid((NN + 15) / 16, 512 / 256);
        k_gemm_wmma_f16<512><<<grid, 128, 0, stream>>>(aggh, WtFF, bff, out, NN, 512);
    }
}
